// SelfAttention_23502061044439
// MI455X (gfx1250) — matrix operand order repaired and checked by reading, not compiled, not run on hardware
//
#include <hip/hip_runtime.h>

// ---------------------------------------------------------------------------
// Multi-head self-attention (B=4, S=2048, H=16, Dk=64, Dmodel=1024)
// bf16 WMMA (v_wmma_f32_16x16x32_bf16) with f32 accumulation, flash-style
// online softmax. wave32 / gfx1250. All fragments loaded as 128-bit chunks;
// GEMM K-loops and attention s-loop are software-pipelined (prefetch next
// fragments before consuming current ones).
// ---------------------------------------------------------------------------

typedef __attribute__((ext_vector_type(16))) __bf16 v16bf;
typedef __attribute__((ext_vector_type(8)))  float  v8f;

#define B_   4
#define S_   2048
#define H_   16
#define DK_  64
#define DM_  1024
#define ROWS_ (B_ * S_)          // 8192

union Frag16 { uint4 x[2]; unsigned u[8]; v16bf v; };
union FragC  { v8f v; float f[8]; };

__device__ __forceinline__ unsigned short f2bf(float f) {
  unsigned u = __float_as_uint(f);
  u += 0x7FFFu + ((u >> 16) & 1u);      // round-to-nearest-even
  return (unsigned short)(u >> 16);
}

__device__ __forceinline__ v8f wmma_bf16(const Frag16& a, const Frag16& b, v8f c) {
  return __builtin_amdgcn_wmma_f32_16x16x32_bf16(
      /*neg_a=*/false, a.v, /*neg_b=*/false, b.v,
      /*c_mod=*/(short)0, c, /*reuse_a=*/false, /*reuse_b=*/false);
}

// ---------------------------------------------------------------------------
// (1) elementwise f32 -> bf16
// ---------------------------------------------------------------------------
__global__ void sa_cvt_bf16(const float* __restrict__ in,
                            unsigned short* __restrict__ out, int n) {
  int i = blockIdx.x * blockDim.x + threadIdx.x;
  int stride = gridDim.x * blockDim.x;
  for (; i < n; i += stride) out[i] = f2bf(in[i]);
}

// ---------------------------------------------------------------------------
// (2) WQ/WK/WV [h][d][n] f32  ->  wt [m][h][n][d] bf16
// ---------------------------------------------------------------------------
__global__ void sa_cvt_wqkv(const float* __restrict__ WQ,
                            const float* __restrict__ WK,
                            const float* __restrict__ WV,
                            unsigned short* __restrict__ wt) {
  int i = blockIdx.x * blockDim.x + threadIdx.x;
  int stride = gridDim.x * blockDim.x;
  const int n_total = 3 * H_ * DK_ * DM_;       // 3 * 2^20
  for (; i < n_total; i += stride) {
    int d = i & (DM_ - 1);
    int n = (i >> 10) & (DK_ - 1);
    int h = (i >> 16) & (H_ - 1);
    int m = i >> 20;
    const float* W = (m == 0) ? WQ : ((m == 1) ? WK : WV);
    wt[i] = f2bf(W[(h * DM_ + d) * DK_ + n]);
  }
}

// ---------------------------------------------------------------------------
// (3) WO [k][n] f32 -> wo_t [n][k] bf16
// ---------------------------------------------------------------------------
__global__ void sa_cvt_wo(const float* __restrict__ WO,
                          unsigned short* __restrict__ wo_t) {
  int i = blockIdx.x * blockDim.x + threadIdx.x;
  int stride = gridDim.x * blockDim.x;
  for (; i < DM_ * DM_; i += stride) {
    int k = i & (DM_ - 1);
    int n = i >> 10;
    wo_t[i] = f2bf(WO[k * DM_ + n]);
  }
}

// ---------------------------------------------------------------------------
// (4) QKV projections. One wave = one (m, head, 16-row tile) -> 16x64 tile.
//     Double-buffered K loop: prefetch k+1 fragments before wmma on k.
// ---------------------------------------------------------------------------
__global__ void __launch_bounds__(256)
sa_qkv_gemm(const unsigned short* __restrict__ xb,
            const unsigned short* __restrict__ wt,
            unsigned short* __restrict__ qb,
            unsigned short* __restrict__ kb,
            unsigned short* __restrict__ vt) {
  const int lane = threadIdx.x & 31;
  const int wid  = blockIdx.x * (blockDim.x >> 5) + (threadIdx.x >> 5);
  const int mt   = wid & 511;           // 512 row tiles of 16
  const int h    = (wid >> 9) & 15;
  const int m    = wid >> 13;           // 0=Q 1=K 2=V
  const int row0 = mt * 16;
  const int lrow = lane & 15;
  const int half = lane >> 4;

  const uint4* xq = (const uint4*)xb;                                   // 8 bf16 / uint4
  const uint4* wq = (const uint4*)(wt + (size_t)(m * H_ + h) * DK_ * DM_);

  // uint4 base indices (bf16 idx / 8)
  const size_t abase = (size_t)(row0 + lrow) * 128 + half * 2;          // + k0/8 ; +1
  const size_t bb0   = (size_t)(0 * 16 + lrow) * 128 + half * 2;        // + k0/8 ; +1
  const size_t bb1   = (size_t)(1 * 16 + lrow) * 128 + half * 2;
  const size_t bb2   = (size_t)(2 * 16 + lrow) * 128 + half * 2;
  const size_t bb3   = (size_t)(3 * 16 + lrow) * 128 + half * 2;

  auto load_all = [&](int k8, Frag16& a, Frag16& b0, Frag16& b1,
                      Frag16& b2, Frag16& b3) {
    a.x[0]  = xq[abase + k8];  a.x[1]  = xq[abase + k8 + 1];
    b0.x[0] = wq[bb0 + k8];    b0.x[1] = wq[bb0 + k8 + 1];
    b1.x[0] = wq[bb1 + k8];    b1.x[1] = wq[bb1 + k8 + 1];
    b2.x[0] = wq[bb2 + k8];    b2.x[1] = wq[bb2 + k8 + 1];
    b3.x[0] = wq[bb3 + k8];    b3.x[1] = wq[bb3 + k8 + 1];
  };

  v8f c0 = {}, c1 = {}, c2 = {}, c3 = {};
  Frag16 a, b0, b1, b2, b3;
  load_all(0, a, b0, b1, b2, b3);
#pragma unroll 4
  for (int k0 = 0; k0 < DM_ - 32; k0 += 32) {
    Frag16 an, b0n, b1n, b2n, b3n;
    load_all((k0 + 32) >> 3, an, b0n, b1n, b2n, b3n);   // prefetch next
    c0 = wmma_bf16(a, b0, c0);
    c1 = wmma_bf16(a, b1, c1);
    c2 = wmma_bf16(a, b2, c2);
    c3 = wmma_bf16(a, b3, c3);
    a = an; b0 = b0n; b1 = b1n; b2 = b2n; b3 = b3n;
  }
  c0 = wmma_bf16(a, b0, c0);
  c1 = wmma_bf16(a, b1, c1);
  c2 = wmma_bf16(a, b2, c2);
  c3 = wmma_bf16(a, b3, c3);

  FragC cc[4];
  cc[0].v = c0; cc[1].v = c1; cc[2].v = c2; cc[3].v = c3;

  const int bidx = row0 >> 11;               // row0 / 2048
  const int s0   = row0 & (S_ - 1);
  const size_t base_qk = (size_t)(bidx * H_ + h) * S_ * DK_;
  const size_t base_v  = (size_t)(bidx * H_ + h) * DK_ * S_;
  unsigned short* dst = (m == 0) ? qb : ((m == 1) ? kb : vt);

#pragma unroll
  for (int t = 0; t < 4; ++t) {
    int n = t * 16 + lrow;
#pragma unroll
    for (int r = 0; r < 8; ++r) {
      int s = s0 + r + 8 * half;             // C layout row
      size_t addr = (m == 2) ? (base_v + (size_t)n * S_ + s)
                             : (base_qk + (size_t)s * DK_ + n);
      dst[addr] = f2bf(cc[t].f[r]);
    }
  }
}

// ---------------------------------------------------------------------------
// (5) Flash attention: one wave = one (b, h, 16-query tile). Pipelined:
//     next-block K and current-block V loads issue before the softmax VALU
//     section; scores-wmma consumes K loaded one iteration earlier.
// ---------------------------------------------------------------------------
__global__ void __launch_bounds__(256)
sa_attn(const unsigned short* __restrict__ qb,
        const unsigned short* __restrict__ kb,
        const unsigned short* __restrict__ vt,
        unsigned short* __restrict__ ob) {
  __shared__ unsigned short pbuf[8][16 * 32];   // per-wave P tile (1 KB each)

  const int lane  = threadIdx.x & 31;
  const int wslot = threadIdx.x >> 5;
  const int wid   = blockIdx.x * 8 + wslot;
  const int qt    = wid & 127;
  const int h     = (wid >> 7) & 15;
  const int b     = wid >> 11;
  const int lrow  = lane & 15;
  const int half  = lane >> 4;
  const float scale = 0.125f;                   // 1/sqrt(64)

  const uint4* q4 = (const uint4*)qb;
  const uint4* k4 = (const uint4*)kb;
  const uint4* v4 = (const uint4*)vt;

  const size_t qk4 = ((size_t)(b * H_ + h) * S_ * DK_) >> 3;   // uint4 units
  const size_t vb4 = ((size_t)(b * H_ + h) * DK_ * S_) >> 3;

  // Q A-fragments: dk 0..31 (qf0) and 32..63 (qf1)
  Frag16 qf0, qf1;
  {
    const size_t qi = qk4 + (size_t)(qt * 16 + lrow) * 8 + half * 2;
    qf0.x[0] = q4[qi];     qf0.x[1] = q4[qi + 1];
    qf1.x[0] = q4[qi + 4]; qf1.x[1] = q4[qi + 5];
  }

  FragC O[4];
#pragma unroll
  for (int t = 0; t < 4; ++t) O[t].v = (v8f){};
  float m_i[8], l_i[8];
#pragma unroll
  for (int r = 0; r < 8; ++r) { m_i[r] = -1e30f; l_i[r] = 0.f; }

  // per-wave loop-invariant bases
  const size_t kbase = qk4 + (size_t)lrow * 8 + half * 2;     // + skey*8
  const size_t vbase = vb4 + (size_t)lrow * 256 + half * 2;   // + t*4096 + s0/8
  unsigned short* pb = pbuf[wslot];
  const uint4* pu4 = (const uint4*)pb;
  const size_t pidx = (size_t)lrow * 4 + half * 2;

  // kf[0/1] = s-tile 0 dk-halves, kf[2/3] = s-tile 1 dk-halves
  auto load_k = [&](int s0, Frag16 kf[4]) {
    const size_t ki0 = kbase + (size_t)s0 * 8;
    const size_t ki1 = kbase + (size_t)(s0 + 16) * 8;
    kf[0].x[0] = k4[ki0];     kf[0].x[1] = k4[ki0 + 1];
    kf[1].x[0] = k4[ki0 + 4]; kf[1].x[1] = k4[ki0 + 5];
    kf[2].x[0] = k4[ki1];     kf[2].x[1] = k4[ki1 + 1];
    kf[3].x[0] = k4[ki1 + 4]; kf[3].x[1] = k4[ki1 + 5];
  };

  Frag16 kc[4];
  load_k(0, kc);

  for (int s0 = 0; s0 < S_; s0 += 32) {
    // ---- prefetch next K block (final iteration over-reads 4 KB into the
    //      adjacent ws buffer; values unused) and this block's V ----
    Frag16 kn[4];
    load_k(s0 + 32, kn);
    Frag16 vf[4];
#pragma unroll
    for (int t = 0; t < 4; ++t) {
      const size_t vi = vbase + (size_t)t * 4096 + (s0 >> 3);
      vf[t].x[0] = v4[vi];
      vf[t].x[1] = v4[vi + 1];
    }

    // ---- scores: two 16x16 f32 tiles over 32 keys (uses kc) ----
    FragC sc[2];
    {
      v8f acc = {};
      acc = wmma_bf16(qf0, kc[0], acc);
      acc = wmma_bf16(qf1, kc[1], acc);
      sc[0].v = acc;
    }
    {
      v8f acc = {};
      acc = wmma_bf16(qf0, kc[2], acc);
      acc = wmma_bf16(qf1, kc[3], acc);
      sc[1].v = acc;
    }

    // ---- online softmax (rows on VGPR index; cols on lanes) ----
    float corrv[8];
#pragma unroll
    for (int r = 0; r < 8; ++r) {
      float a0 = sc[0].f[r] * scale;
      float a1 = sc[1].f[r] * scale;
      float t0 = fmaxf(a0, a1);
      t0 = fmaxf(t0, __shfl_xor(t0, 1));
      t0 = fmaxf(t0, __shfl_xor(t0, 2));
      t0 = fmaxf(t0, __shfl_xor(t0, 4));
      t0 = fmaxf(t0, __shfl_xor(t0, 8));
      float mnew = fmaxf(m_i[r], t0);
      float corr = __expf(m_i[r] - mnew);
      m_i[r] = mnew;
      float p0 = __expf(a0 - mnew);
      float p1 = __expf(a1 - mnew);
      sc[0].f[r] = p0;
      sc[1].f[r] = p1;
      float rs = p0 + p1;
      rs += __shfl_xor(rs, 1);
      rs += __shfl_xor(rs, 2);
      rs += __shfl_xor(rs, 4);
      rs += __shfl_xor(rs, 8);
      l_i[r] = l_i[r] * corr + rs;
      corrv[r] = corr;
    }
#pragma unroll
    for (int r = 0; r < 8; ++r) {
      O[0].f[r] *= corrv[r]; O[1].f[r] *= corrv[r];
      O[2].f[r] *= corrv[r]; O[3].f[r] *= corrv[r];
    }

    // ---- P: C-frag layout -> LDS -> A-frag layout (bf16) ----
#pragma unroll
    for (int r = 0; r < 8; ++r) {
      int row = r + 8 * half;
      pb[row * 32 + lrow]      = f2bf(sc[0].f[r]);
      pb[row * 32 + 16 + lrow] = f2bf(sc[1].f[r]);
    }
    asm volatile("s_wait_dscnt 0" ::: "memory");
    Frag16 pa;
    pa.x[0] = pu4[pidx];
    pa.x[1] = pu4[pidx + 1];

    // ---- O += P @ V ----
#pragma unroll
    for (int t = 0; t < 4; ++t) O[t].v = wmma_bf16(pa, vf[t], O[t].v);

    // rotate pipeline
#pragma unroll
    for (int t = 0; t < 4; ++t) kc[t] = kn[t];
  }

  // ---- epilogue: normalize and store concat-head bf16 O ----
  const size_t obase = ((size_t)b * S_ + (size_t)qt * 16) * DM_ + (size_t)h * DK_;
#pragma unroll
  for (int r = 0; r < 8; ++r) {
    float inv = 1.f / l_i[r];
    int Mrow = r + 8 * half;
#pragma unroll
    for (int t = 0; t < 4; ++t) {
      int n = t * 16 + lrow;
      ob[obase + (size_t)Mrow * DM_ + n] = f2bf(O[t].f[r] * inv);
    }
  }
}

// ---------------------------------------------------------------------------
// (6) Output projection: [8192,1024]bf16 @ WO^T[n][k] -> f32 out.
//     Same double-buffered structure as the QKV GEMM.
// ---------------------------------------------------------------------------
__global__ void __launch_bounds__(256)
sa_out_gemm(const unsigned short* __restrict__ ob,
            const unsigned short* __restrict__ wo_t,
            float* __restrict__ out) {
  const int lane = threadIdx.x & 31;
  const int wid  = blockIdx.x * 8 + (threadIdx.x >> 5);
  const int nt   = wid & 15;            // 16 n-tiles of 64
  const int mt   = wid >> 4;            // 512 row tiles of 16
  const int row0 = mt * 16;
  const int n0   = nt * 64;
  const int lrow = lane & 15;
  const int half = lane >> 4;

  const uint4* oq = (const uint4*)ob;
  const uint4* wq = (const uint4*)wo_t;

  const size_t abase = (size_t)(row0 + lrow) * 128 + half * 2;        // + k0/8 ; +1
  const size_t bb0   = (size_t)(n0 + 0 * 16 + lrow) * 128 + half * 2; // + k0/8 ; +1
  const size_t bb1   = (size_t)(n0 + 1 * 16 + lrow) * 128 + half * 2;
  const size_t bb2   = (size_t)(n0 + 2 * 16 + lrow) * 128 + half * 2;
  const size_t bb3   = (size_t)(n0 + 3 * 16 + lrow) * 128 + half * 2;

  auto load_all = [&](int k8, Frag16& a, Frag16& b0, Frag16& b1,
                      Frag16& b2, Frag16& b3) {
    a.x[0]  = oq[abase + k8];  a.x[1]  = oq[abase + k8 + 1];
    b0.x[0] = wq[bb0 + k8];    b0.x[1] = wq[bb0 + k8 + 1];
    b1.x[0] = wq[bb1 + k8];    b1.x[1] = wq[bb1 + k8 + 1];
    b2.x[0] = wq[bb2 + k8];    b2.x[1] = wq[bb2 + k8 + 1];
    b3.x[0] = wq[bb3 + k8];    b3.x[1] = wq[bb3 + k8 + 1];
  };

  v8f c0 = {}, c1 = {}, c2 = {}, c3 = {};
  Frag16 a, b0, b1, b2, b3;
  load_all(0, a, b0, b1, b2, b3);
#pragma unroll 4
  for (int k0 = 0; k0 < DM_ - 32; k0 += 32) {
    Frag16 an, b0n, b1n, b2n, b3n;
    load_all((k0 + 32) >> 3, an, b0n, b1n, b2n, b3n);   // prefetch next
    c0 = wmma_bf16(a, b0, c0);
    c1 = wmma_bf16(a, b1, c1);
    c2 = wmma_bf16(a, b2, c2);
    c3 = wmma_bf16(a, b3, c3);
    a = an; b0 = b0n; b1 = b1n; b2 = b2n; b3 = b3n;
  }
  c0 = wmma_bf16(a, b0, c0);
  c1 = wmma_bf16(a, b1, c1);
  c2 = wmma_bf16(a, b2, c2);
  c3 = wmma_bf16(a, b3, c3);

  FragC cc[4];
  cc[0].v = c0; cc[1].v = c1; cc[2].v = c2; cc[3].v = c3;
#pragma unroll
  for (int t = 0; t < 4; ++t) {
#pragma unroll
    for (int r = 0; r < 8; ++r) {
      out[(size_t)(row0 + r + 8 * half) * DM_ + n0 + t * 16 + lrow] = cc[t].f[r];
    }
  }
}

// ---------------------------------------------------------------------------
extern "C" void kernel_launch(void* const* d_in, const int* in_sizes, int n_in,
                              void* d_out, int out_size, void* d_ws, size_t ws_size,
                              hipStream_t stream) {
  (void)in_sizes; (void)n_in; (void)out_size; (void)ws_size;
  const float* x  = (const float*)d_in[0];
  const float* WQ = (const float*)d_in[1];
  const float* WK = (const float*)d_in[2];
  const float* WV = (const float*)d_in[3];
  const float* WO = (const float*)d_in[4];
  float* out = (float*)d_out;

  char* ws = (char*)d_ws;
  size_t off = 0;
  unsigned short* xb   = (unsigned short*)(ws + off); off += (size_t)ROWS_ * DM_ * 2;        // 16 MB
  unsigned short* wt   = (unsigned short*)(ws + off); off += (size_t)3 * H_ * DK_ * DM_ * 2; //  6 MB
  unsigned short* wo_t = (unsigned short*)(ws + off); off += (size_t)DM_ * DM_ * 2;          //  2 MB
  unsigned short* qb   = (unsigned short*)(ws + off); off += (size_t)B_ * H_ * S_ * DK_ * 2; // 16 MB
  unsigned short* kb   = (unsigned short*)(ws + off); off += (size_t)B_ * H_ * S_ * DK_ * 2; // 16 MB
  unsigned short* vt   = (unsigned short*)(ws + off); off += (size_t)B_ * H_ * S_ * DK_ * 2; // 16 MB
  unsigned short* ob   = (unsigned short*)(ws + off); off += (size_t)ROWS_ * DM_ * 2;        // 16 MB

  sa_cvt_bf16<<<2048, 256, 0, stream>>>(x, xb, ROWS_ * DM_);
  sa_cvt_wqkv<<<2048, 256, 0, stream>>>(WQ, WK, WV, wt);
  sa_cvt_wo  <<<1024, 256, 0, stream>>>(WO, wo_t);
  sa_qkv_gemm<<<3072, 256, 0, stream>>>(xb, wt, qb, kb, vt);
  sa_attn    <<<1024, 256, 0, stream>>>(qb, kb, vt, ob);
  sa_out_gemm<<<1024, 256, 0, stream>>>(ob, wo_t, out);
}
